// AttentionMIL_79044578116031
// MI455X (gfx1250) — matrix operand order repaired and checked by reading, not compiled, not run on hardware
//
#include <hip/hip_runtime.h>
#include <math.h>

typedef __attribute__((ext_vector_type(16))) __bf16 v16bf;
typedef __attribute__((ext_vector_type(8)))  __bf16 v8bf;
typedef __attribute__((ext_vector_type(8)))  float  v8f;
typedef int i32x4 __attribute__((__vector_size__(16)));

#define B_ 8
#define T_ 8192
#define F_ 1024
#define H_ 256

#if __has_builtin(__builtin_amdgcn_global_load_async_to_lds_b128)
#define HAVE_ASYNC_LDS 1
#else
#define HAVE_ASYNC_LDS 0
#endif
#if __has_builtin(__builtin_amdgcn_global_store_async_from_lds_b128)
#define HAVE_ASYNC_ST 1
#else
#define HAVE_ASYNC_ST 0
#endif

static __device__ __forceinline__ void sched_fence() {
#if __has_builtin(__builtin_amdgcn_sched_barrier)
  __builtin_amdgcn_sched_barrier(0);  // nothing may be reordered across
#endif
}

typedef __attribute__((address_space(1))) i32x4* gvec_p;   // global int4*
typedef __attribute__((address_space(3))) i32x4* lvec_p;   // LDS int4*

static __device__ __forceinline__ lvec_p to_lds(const void* p) {
  // low 32 bits of the flat address of a __shared__ object = LDS byte offset
  return (lvec_p)(unsigned)(uintptr_t)p;
}
static __device__ __forceinline__ gvec_p to_glob(const void* p) {
  return (gvec_p)(uintptr_t)p;
}

// 16B global -> LDS, async (ASYNCcnt) when available, else synchronous.
static __device__ __forceinline__ void copy16_g2l(const void* g, void* l) {
#if HAVE_ASYNC_LDS
  __builtin_amdgcn_global_load_async_to_lds_b128(to_glob(g), to_lds(l), 0, 0);
#else
  *(uint4*)l = *(const uint4*)g;
#endif
}

// 16B LDS -> global, async when available, else synchronous.
static __device__ __forceinline__ void copy16_l2g(void* g, const void* l) {
#if HAVE_ASYNC_ST
  __builtin_amdgcn_global_store_async_from_lds_b128(to_glob(g), to_lds(l), 0, 0);
#else
  *(uint4*)g = *(const uint4*)l;
#endif
}

static __device__ __forceinline__ void wait_async_le4() {
#if HAVE_ASYNC_LDS
#if __has_builtin(__builtin_amdgcn_s_wait_asynccnt)
  __builtin_amdgcn_s_wait_asynccnt(4);
#else
  asm volatile("s_wait_asynccnt 4" ::: "memory");
#endif
#endif
}
static __device__ __forceinline__ void wait_async_le0() {
#if HAVE_ASYNC_LDS
#if __has_builtin(__builtin_amdgcn_s_wait_asynccnt)
  __builtin_amdgcn_s_wait_asynccnt(0);
#else
  asm volatile("s_wait_asynccnt 0" ::: "memory");
#endif
#endif
}

static __device__ __forceinline__ unsigned short f2bf_bits(float f) {
  __bf16 h = (__bf16)f;
  return __builtin_bit_cast(unsigned short, h);
}

static __device__ __forceinline__ v8f zero8() {
  v8f z;
#pragma unroll
  for (int i = 0; i < 8; ++i) z[i] = 0.0f;
  return z;
}

// ---------------------------------------------------------------------------
// Pass 0: f32 K x N weight -> bf16 tiles in this file's B-fragment lane order (contiguous k).
// Tile = 32(K) x 16(N); element index = L*16 + e with
//   N = nt*16 + L%16 ; K = kt*32 + (L/16)*16 + e.
// ---------------------------------------------------------------------------
__global__ void swizzle_b_kernel(const float* __restrict__ src,
                                 __bf16* __restrict__ dst, int K, int N) {
  int tid = blockIdx.x * blockDim.x + threadIdx.x;
  if (tid >= K * N) return;
  int tile = tid >> 9;
  int r    = tid & 511;
  int L    = r >> 4;
  int e    = r & 15;
  int ntn  = N >> 4;
  int kt   = tile / ntn;
  int nt   = tile % ntn;
  int k    = kt * 32 + (L >> 4) * 16 + e;
  int n    = nt * 16 + (L & 15);
  dst[tid] = (__bf16)src[(size_t)k * N + n];
}

// ---------------------------------------------------------------------------
// Pass 1 (fused): h = relu(bags@W1+b1); gated-attention logit per t; h spilled
// row-major bf16 to ws via async LDS->global stores.
// 256 threads = 8 waves; each wave owns 16 rows. W1's per-K-step 16KB B-tile
// is staged into LDS once per block (async, double buffered).
// ---------------------------------------------------------------------------
__global__ __launch_bounds__(256) void mil_pass1(
    const float* __restrict__ bags,
    const __bf16* __restrict__ wB1,
    const __bf16* __restrict__ wBv,
    const __bf16* __restrict__ wBu,
    const float* __restrict__ b1,
    const float* __restrict__ bv,
    const float* __restrict__ bu,
    const float* __restrict__ ww,
    const float* __restrict__ bw,
    __bf16* __restrict__ h_ws,        // [B*T][H] bf16, row-major
    float* __restrict__ logits_ws) {  // [B*T]
  // 0..32KB: W1 B-tile double buffer during GEMM1.
  // Reused afterwards (full 64KB) as 8 waves x (16 rows x 256) bf16 h staging.
  __shared__ __attribute__((aligned(32))) unsigned char ldsRaw[65536];

  const int tid  = threadIdx.x;
  const int lane = tid & 31;
  const int wave = tid >> 5;
  const int m16  = lane & 15;
  const int hh   = lane >> 4;
  const int g0   = blockIdx.x * 128 + wave * 16;  // global row base (b*T+t)

  // ---- stage helper: copy 16KB tile (W1, K-step ks) into LDS buffer ----
  const unsigned char* wB1b = (const unsigned char*)wB1;
#define STAGE_B(bufsel, ks)                                              \
  {                                                                      \
    const unsigned char* g = wB1b + ((size_t)(ks) << 14) + (tid << 4);   \
    unsigned char* l = ldsRaw + ((bufsel) << 14) + (tid << 4);           \
    _Pragma("unroll") for (int r = 0; r < 4; ++r)                        \
        copy16_g2l(g + r * 4096, l + r * 4096);                          \
  }

  // ---- GEMM1: 16 x 256 strip, K = 1024, bf16 WMMA ----
  v8f acc[16];
#pragma unroll
  for (int i = 0; i < 16; ++i) acc[i] = zero8();

  const float* aBase = bags + (size_t)(g0 + m16) * F_ + hh * 16;

  STAGE_B(0, 0);
  for (int ks = 0; ks < 32; ++ks) {
    if (ks < 31) {
      STAGE_B((ks + 1) & 1, ks + 1);
      wait_async_le4();  // previous stage complete (loads retire in order)
    } else {
      wait_async_le0();
    }
    __syncthreads();  // B tile for this ks visible to all waves

    const float* ap = aBase + ks * 32;
    __builtin_prefetch(ap + 64, 0, 0);  // global_prefetch next A chunk
    float4 c00 = *(const float4*)(ap);
    float4 c01 = *(const float4*)(ap + 4);
    float4 c10 = *(const float4*)(ap + 8);
    float4 c11 = *(const float4*)(ap + 12);
    v16bf a;
    a[0]  = (__bf16)c00.x; a[1]  = (__bf16)c00.y; a[2]  = (__bf16)c00.z; a[3]  = (__bf16)c00.w;
    a[4]  = (__bf16)c01.x; a[5]  = (__bf16)c01.y; a[6]  = (__bf16)c01.z; a[7]  = (__bf16)c01.w;
    a[8]  = (__bf16)c10.x; a[9]  = (__bf16)c10.y; a[10] = (__bf16)c10.z; a[11] = (__bf16)c10.w;
    a[12] = (__bf16)c11.x; a[13] = (__bf16)c11.y; a[14] = (__bf16)c11.z; a[15] = (__bf16)c11.w;

    const unsigned char* bbase = ldsRaw + ((ks & 1) << 14) + (lane << 5);
    // 2-deep software pipeline on B fragments: loads for nt+2 are issued
    // before the WMMA for nt; sched_fence() keeps the compiler from
    // collapsing the pipeline (wait becomes dscnt<=2, not 0).
    v16bf bcur = *(const v16bf*)(bbase);
    v16bf bnxt = *(const v16bf*)(bbase + (1 << 10));
#pragma unroll
    for (int nt = 0; nt < 16; ++nt) {
      v16bf bfut = bnxt;
      if (nt < 14) bfut = *(const v16bf*)(bbase + ((nt + 2) << 10));
      sched_fence();
      acc[nt] = __builtin_amdgcn_wmma_f32_16x16x32_bf16(
          false, a, false, bcur, (short)0, acc[nt], false, false);
      bcur = bnxt;
      bnxt = bfut;
    }
    __syncthreads();  // all waves done reading before buffer is overwritten
  }

  // ---- bias + ReLU; stage h into LDS row-major (wave-private 8KB) ----
  unsigned short* myLds = (unsigned short*)(ldsRaw + wave * 8192);
#pragma unroll
  for (int nt = 0; nt < 16; ++nt) {
    const int n   = nt * 16 + m16;
    const float bb = b1[n];
#pragma unroll
    for (int r = 0; r < 8; ++r) {
      float v = acc[nt][r] + bb;
      myLds[(r + 8 * hh) * 256 + n] = f2bf_bits(v > 0.0f ? v : 0.0f);
    }
  }
  __syncthreads();  // h staging committed (drains DS stores)

  // ---- spill h to global row-major via async LDS->global (contiguous) ----
  {
    unsigned char* gdst = (unsigned char*)h_ws + (size_t)g0 * 512;
    unsigned char* lsrc = ldsRaw + wave * 8192;
#pragma unroll
    for (int r = 0; r < 16; ++r)
      copy16_l2g(gdst + (lane << 4) + r * 512, lsrc + (lane << 4) + r * 512);
  }

  // ---- GEMM2/3 per N-tile: a_v = tanh(h@Wv+bv), a_u = sigm(h@Wu+bu),
  //      logit += (a_v * a_u) . ww ----
  float logit[8];
#pragma unroll
  for (int r = 0; r < 8; ++r) logit[r] = 0.0f;

  for (int nt = 0; nt < 16; ++nt) {
    v8f vacc = zero8(), uacc = zero8();
    size_t t0 = ((size_t)nt << 9) + (lane << 4);
    v16bf bvc = *(const v16bf*)(wBv + t0);
    v16bf buc = *(const v16bf*)(wBu + t0);
#pragma unroll
    for (int ks = 0; ks < 8; ++ks) {
      v16bf bvn = bvc, bun = buc;
      if (ks < 7) {
        size_t t1 = ((size_t)((ks + 1) * 16 + nt) << 9) + (lane << 4);
        bvn = *(const v16bf*)(wBv + t1);
        bun = *(const v16bf*)(wBu + t1);
      }
      const unsigned short* lp = myLds + m16 * 256 + ks * 32 + hh * 16;
      v8bf lo = *(const v8bf*)(lp);
      v8bf hi = *(const v8bf*)(lp + 8);
      v16bf a;
#pragma unroll
      for (int i = 0; i < 8; ++i) { a[i] = lo[i]; a[8 + i] = hi[i]; }
      sched_fence();  // keep next-ks fragment loads ahead of these WMMAs
      vacc = __builtin_amdgcn_wmma_f32_16x16x32_bf16(
          false, a, false, bvc, (short)0, vacc, false, false);
      uacc = __builtin_amdgcn_wmma_f32_16x16x32_bf16(
          false, a, false, buc, (short)0, uacc, false, false);
      bvc = bvn; buc = bun;
    }
    const int n = nt * 16 + m16;
    const float bvv = bv[n], buu = bu[n], wwn = ww[n];
#pragma unroll
    for (int r = 0; r < 8; ++r) {
      float g = tanhf(vacc[r] + bvv) *
                (1.0f / (1.0f + __expf(-(uacc[r] + buu))));
      logit[r] += g * wwn;
    }
  }

  // C-layout: lanes 0-15 hold row r, lanes 16-31 hold row r+8.
#pragma unroll
  for (int r = 0; r < 8; ++r) {
    float v = logit[r];
    v += __shfl_xor(v, 1, 16);
    v += __shfl_xor(v, 2, 16);
    v += __shfl_xor(v, 4, 16);
    v += __shfl_xor(v, 8, 16);
    logit[r] = v;
  }
  if (m16 == 0) {
    const float bw0 = bw[0];
#pragma unroll
    for (int r = 0; r < 8; ++r) logits_ws[g0 + 8 * hh + r] = logit[r] + bw0;
  }
#undef STAGE_B
}

// ---------------------------------------------------------------------------
// Pass 2: softmax over T per bag (mask -> -inf).
// ---------------------------------------------------------------------------
__global__ __launch_bounds__(256) void mil_softmax(
    const float* __restrict__ logits, const unsigned char* __restrict__ mask,
    float* __restrict__ attn) {
  __shared__ float red[256];
  const int b = blockIdx.x;
  const int tid = threadIdx.x;
  const float* lp = logits + (size_t)b * T_;
  const unsigned char* mp = mask + (size_t)b * T_;
  float mx = -INFINITY;
  for (int i = tid; i < T_; i += 256) {
    float l = mp[i] ? -INFINITY : lp[i];
    mx = fmaxf(mx, l);
  }
  red[tid] = mx; __syncthreads();
  for (int s = 128; s > 0; s >>= 1) {
    if (tid < s) red[tid] = fmaxf(red[tid], red[tid + s]);
    __syncthreads();
  }
  mx = red[0]; __syncthreads();
  float sum = 0.0f;
  for (int i = tid; i < T_; i += 256) {
    float l = mp[i] ? -INFINITY : lp[i];
    sum += __expf(l - mx);
  }
  red[tid] = sum; __syncthreads();
  for (int s = 128; s > 0; s >>= 1) {
    if (tid < s) red[tid] += red[tid + s];
    __syncthreads();
  }
  const float inv = 1.0f / red[0];
  for (int i = tid; i < T_; i += 256) {
    float l = mp[i] ? -INFINITY : lp[i];
    attn[(size_t)b * T_ + i] = __expf(l - mx) * inv;
  }
}

// ---------------------------------------------------------------------------
// Pass 3: partial weighted aggregation over t-chunks (coalesced, row-major h).
// block = (b, tc); thread owns one hidden channel n.
// ---------------------------------------------------------------------------
__global__ __launch_bounds__(256) void mil_pass3(
    const __bf16* __restrict__ h_ws, const float* __restrict__ attn,
    float* __restrict__ partial) {
  const int b  = blockIdx.x >> 4;
  const int tc = blockIdx.x & 15;
  const int n  = threadIdx.x;
  const __bf16* hp = h_ws + ((size_t)b * T_ + (size_t)tc * 512) * H_ + n;
  const float* ap = attn + (size_t)b * T_ + tc * 512;
  float acc = 0.0f;
  for (int t = 0; t < 512; ++t) acc += ap[t] * (float)hp[(size_t)t * H_];
  partial[(b * 16 + tc) * H_ + n] = acc;
}

__global__ __launch_bounds__(256) void mil_reduce(
    const float* __restrict__ partial, float* __restrict__ slide) {
  int i = blockIdx.x * 256 + threadIdx.x;  // 0..2047
  int b = i >> 8, n = i & 255;
  float s = 0.0f;
  for (int tc = 0; tc < 16; ++tc) s += partial[(b * 16 + tc) * H_ + n];
  slide[i] = s;
}

// ---------------------------------------------------------------------------
// Pass 4: logits = slide_repr @ Wc + bc  (8 x 256 x 2, trivial).
// ---------------------------------------------------------------------------
__global__ void mil_head(const float* __restrict__ slide,
                         const float* __restrict__ Wc,
                         const float* __restrict__ bc,
                         float* __restrict__ out) {
  int i = threadIdx.x;
  if (i >= B_ * 2) return;
  int b = i >> 1, o = i & 1;
  float s = bc[o];
  for (int n = 0; n < H_; ++n) s += slide[b * H_ + n] * Wc[n * 2 + o];
  out[i] = s;
}

extern "C" void kernel_launch(void* const* d_in, const int* in_sizes, int n_in,
                              void* d_out, int out_size, void* d_ws,
                              size_t ws_size, hipStream_t stream) {
  const float* bags          = (const float*)d_in[0];
  const unsigned char* mask  = (const unsigned char*)d_in[1];
  const float* W1            = (const float*)d_in[2];
  const float* b1            = (const float*)d_in[3];
  const float* Wv            = (const float*)d_in[4];
  const float* bvv           = (const float*)d_in[5];
  const float* Wu            = (const float*)d_in[6];
  const float* bu            = (const float*)d_in[7];
  const float* ww            = (const float*)d_in[8];
  const float* bw            = (const float*)d_in[9];
  const float* Wc            = (const float*)d_in[10];
  const float* bc            = (const float*)d_in[11];
  float* out = (float*)d_out;

  char* w = (char*)d_ws;
  __bf16* wsB1 = (__bf16*)w; w += (size_t)F_ * H_ * 2;       // 512 KB
  __bf16* wsBv = (__bf16*)w; w += (size_t)H_ * H_ * 2;       // 128 KB
  __bf16* wsBu = (__bf16*)w; w += (size_t)H_ * H_ * 2;       // 128 KB
  __bf16* h_ws = (__bf16*)w; w += (size_t)B_ * T_ * H_ * 2;  // 32 MB
  float* logits_ws = (float*)w; w += (size_t)B_ * T_ * 4;    // 256 KB
  float* attn_ws   = (float*)w; w += (size_t)B_ * T_ * 4;    // 256 KB
  float* partial_ws = (float*)w; w += (size_t)B_ * 16 * H_ * 4;  // 128 KB
  float* slide_ws  = (float*)w; w += (size_t)B_ * H_ * 4;    // 8 KB

  swizzle_b_kernel<<<(F_ * H_) / 256, 256, 0, stream>>>(W1, wsB1, F_, H_);
  swizzle_b_kernel<<<(H_ * H_) / 256, 256, 0, stream>>>(Wv, wsBv, H_, H_);
  swizzle_b_kernel<<<(H_ * H_) / 256, 256, 0, stream>>>(Wu, wsBu, H_, H_);
  mil_pass1<<<(B_ * T_) / 128, 256, 0, stream>>>(
      bags, wsB1, wsBv, wsBu, b1, bvv, bu, ww, bw, h_ws, logits_ws);
  mil_softmax<<<B_, 256, 0, stream>>>(logits_ws, mask, attn_ws);
  mil_pass3<<<(B_ * 16), 256, 0, stream>>>(h_ws, attn_ws, partial_ws);
  mil_reduce<<<B_, 256, 0, stream>>>(partial_ws, slide_ws);
  mil_head<<<1, 32, 0, stream>>>(slide_ws, Wc, bc, out);
}
